// VQVAEQuantizer_5377299055135
// MI455X (gfx1250) — matrix operand order repaired and checked by reading, not compiled, not run on hardware
//
#include <hip/hip_runtime.h>

typedef __attribute__((ext_vector_type(8)))  _Float16 v8h;
typedef __attribute__((ext_vector_type(16))) _Float16 v16h;
typedef __attribute__((ext_vector_type(8)))  float    v8f;
typedef __attribute__((ext_vector_type(4)))  int      v4i;

#define N_PTS   16384      // B*H*W = 16*32*32
#define K_CODES 8192
#define C_DIM   128
#define HW      1024       // 32*32

// ---- gfx1250 async global->LDS copy (ASYNCcnt path), with safe fallback ----
#if defined(__has_builtin)
#  if __has_builtin(__builtin_amdgcn_global_load_async_to_lds_b128) && \
      __has_builtin(__builtin_amdgcn_s_wait_asynccnt)
#    define USE_ASYNC_LDS 1
#  endif
#endif
#ifndef USE_ASYNC_LDS
#  define USE_ASYNC_LDS 0
#endif

__device__ __forceinline__ void copy16_g2l(_Float16* ldst, const _Float16* gsrc) {
#if USE_ASYNC_LDS
    // probe-confirmed param0 type: int4 in global (__device__/AS1) address space
    __builtin_amdgcn_global_load_async_to_lds_b128(
        (__attribute__((address_space(1))) v4i*)const_cast<_Float16*>(gsrc),
        (__attribute__((address_space(3))) v4i*)ldst,
        /*imm offset*/0, /*cpol*/0);
#else
    *(v8h*)ldst = *(const v8h*)gsrc;
#endif
}

__device__ __forceinline__ void async_fence_to_barrier() {
#if USE_ASYNC_LDS
    __builtin_amdgcn_s_wait_asynccnt(0);
#endif
    __syncthreads();
}

// ---------------------------------------------------------------------------
// Kernel 1: encodings [B,C,H,W] f32 -> xh [N,C] f16 via LDS 32x32 tile
// transpose; both global sides coalesced.
// ---------------------------------------------------------------------------
__global__ __launch_bounds__(256) void vq_prep_x(const float* __restrict__ enc,
                                                 _Float16* __restrict__ xh) {
    __shared__ float tile[32][33];
    const int blk = blockIdx.x;                 // b*(4*32) + (c-tile)*32 + hw-tile
    const int b   = blk >> 7;
    const int c0  = ((blk >> 5) & 3) * 32;
    const int hw0 = (blk & 31) * 32;
    const int tx = threadIdx.x & 31, ty = threadIdx.x >> 5;   // ty: 0..7
    #pragma unroll
    for (int i = 0; i < 4; ++i) {
        int c = c0 + ty + i * 8;
        tile[ty + i * 8][tx] = enc[((size_t)(b * C_DIM + c)) * HW + hw0 + tx];
    }
    __syncthreads();
    #pragma unroll
    for (int i = 0; i < 4; ++i) {
        int hw = hw0 + ty + i * 8;
        xh[((size_t)(b * HW + hw)) * C_DIM + c0 + tx] = (_Float16)tile[tx][ty + i * 8];
    }
}

// ---------------------------------------------------------------------------
// Kernel 2: weight [K,C] f32 -> wh [K,C] f16, wnorm[k] = ||w_k||^2 (f32)
// ---------------------------------------------------------------------------
__global__ __launch_bounds__(128) void vq_prep_w(const float* __restrict__ w,
                                                 _Float16* __restrict__ wh,
                                                 float* __restrict__ wnorm) {
    int k = blockIdx.x, c = threadIdx.x;
    float v = w[(size_t)k * C_DIM + c];
    wh[(size_t)k * C_DIM + c] = (_Float16)v;
    float s = v * v;
    #pragma unroll
    for (int m = 16; m >= 1; m >>= 1) s += __shfl_xor(s, m, 32);
    __shared__ float part[4];
    if ((c & 31) == 0) part[c >> 5] = s;
    __syncthreads();
    if (c == 0) wnorm[k] = part[0] + part[1] + part[2] + part[3];
}

// ---------------------------------------------------------------------------
// Kernel 3: fused GEMM + argmin via v_wmma_f32_16x16x32_f16.
// Wave: 16-row M-tile, A resident in VGPRs for the whole K sweep.
// Block: 8 waves, 32-code double-buffered LDS tile filled by async copies.
// score[m][n] = wnorm[n] - 2*dot(x[m], w[n])   (||x||^2 dropped: row-const)
// ---------------------------------------------------------------------------
__global__ __launch_bounds__(256) void vq_argmin_wmma(const _Float16* __restrict__ xh,
                                                      const _Float16* __restrict__ wh,
                                                      const float* __restrict__ wnorm,
                                                      int* __restrict__ ids) {
    __shared__ _Float16 wt[2][32 * C_DIM];   // 2 x 8 KB weight tiles

    const int tid  = threadIdx.x;
    const int wave = tid >> 5;
    const int lane = tid & 31;
    const int col  = lane & 15;      // column / row-in-tile selector
    const int half = lane >> 4;      // lane half (K split per ISA layouts)
    const int m0   = blockIdx.x * 128 + wave * 16;

    // ---- A fragments: rows m0..m0+15, all C=128 (4 chunks of K=32) ----
    // 16-bit A 16x32: lanes 0-15 hold K={0..7,16..23}; lanes 16-31 K={8..15,24..31}
    v16h a[4];
    const _Float16* xrow = xh + (size_t)(m0 + col) * C_DIM;
    #pragma unroll
    for (int kc = 0; kc < 4; ++kc) {
        v8h lo = *(const v8h*)(xrow + kc * 32 + half * 8);
        v8h hi = *(const v8h*)(xrow + kc * 32 + 16 + half * 8);
        #pragma unroll
        for (int i = 0; i < 8; ++i) { a[kc][i] = lo[i]; a[kc][i + 8] = hi[i]; }
    }

    // cooperative tile copy: 32 rows x 128 f16 = 8 KB = 256 threads x 2 x 16 B
    const int r0 = tid >> 4, c8 = (tid & 15) * 8;
    #define LOAD_TILE(buf, n0)                                                        \
        do {                                                                          \
            copy16_g2l(&wt[buf][r0 * C_DIM + c8],                                     \
                       wh + (size_t)((n0) + r0) * C_DIM + c8);                        \
            copy16_g2l(&wt[buf][(r0 + 16) * C_DIM + c8],                              \
                       wh + (size_t)((n0) + r0 + 16) * C_DIM + c8);                   \
        } while (0)

    LOAD_TILE(0, 0);
    async_fence_to_barrier();

    float best[8];
    int   bidx[8];
    #pragma unroll
    for (int r = 0; r < 8; ++r) { best[r] = 3.4e38f; bidx[r] = 0; }

    const int NT = K_CODES / 32;       // 256 iterations
    for (int t = 0; t < NT; ++t) {
        const int cur = t & 1;
        const int nb  = t * 32;
        if (t + 1 < NT) LOAD_TILE(cur ^ 1, nb + 32);

        const float wn0 = wnorm[nb + col];
        const float wn1 = wnorm[nb + 16 + col];

        #pragma unroll
        for (int s = 0; s < 2; ++s) {
            // B 32x16 layout: lanes 0-15 hold K={0..7,16..23}; lanes 16-31 K={8..15,24..31}
            v16h bfrag[4];
            #pragma unroll
            for (int kc = 0; kc < 4; ++kc) {
                v8h blo = *(const v8h*)(
                    &wt[cur][(s * 16 + col) * C_DIM + kc * 32 + half * 8]);
                v8h bhi = *(const v8h*)(
                    &wt[cur][(s * 16 + col) * C_DIM + kc * 32 + 16 + half * 8]);
                #pragma unroll
                for (int i = 0; i < 8; ++i) { bfrag[kc][i] = blo[i]; bfrag[kc][i + 8] = bhi[i]; }
            }

            v8f acc = {};
            #pragma unroll
            for (int kc = 0; kc < 4; ++kc)
                acc = __builtin_amdgcn_wmma_f32_16x16x32_f16(
                          false, a[kc], false, bfrag[kc], (short)0, acc, false, false);

            const float wn = s ? wn1 : wn0;
            const int   n  = nb + s * 16 + col;
            #pragma unroll
            for (int r = 0; r < 8; ++r) {
                float sc = wn - 2.0f * acc[r];
                if (sc < best[r]) { best[r] = sc; bidx[r] = n; }
            }
        }
        async_fence_to_barrier();
    }
    #undef LOAD_TILE

    // ---- argmin across the 16 columns in each half-wave ----
    #pragma unroll
    for (int m = 1; m < 16; m <<= 1) {
        #pragma unroll
        for (int r = 0; r < 8; ++r) {
            float ob = __shfl_xor(best[r], m, 32);
            int   oi = __shfl_xor(bidx[r], m, 32);
            if (ob < best[r] || (ob == best[r] && oi < bidx[r])) { best[r] = ob; bidx[r] = oi; }
        }
    }
    // C/D layout: VGPR r holds M=r (lanes 0-15) and M=r+8 (lanes 16-31)
    if (col == 0) {
        #pragma unroll
        for (int r = 0; r < 8; ++r) ids[m0 + half * 8 + r] = bidx[r];
    }
}

// ---------------------------------------------------------------------------
// Kernel 4: q = weight[id] scattered to NCHW + EMA stats via atomics.
// Block: 32 consecutive points x all 128 channels -> coalesced q/enc streams.
// ---------------------------------------------------------------------------
__global__ __launch_bounds__(256) void vq_scatter(const int* __restrict__ ids,
                                                  const float* __restrict__ w,
                                                  const float* __restrict__ enc,
                                                  float* __restrict__ q,
                                                  float* __restrict__ sums,
                                                  float* __restrict__ counts) {
    __shared__ int sid[32];
    const int n0 = blockIdx.x * 32;
    const int t  = threadIdx.x;
    if (t < 32) {
        int id = ids[n0 + t];
        sid[t] = id;
        atomicAdd(&counts[id], 1.0f);
    }
    __syncthreads();
    const int tx = t & 31;            // point offset (consecutive hw)
    const int ty = t >> 5;            // channel group (16 channels each)
    const int n  = n0 + tx;
    const int id = sid[tx];
    const int b  = n / HW, hw = n % HW;
    #pragma unroll
    for (int i = 0; i < 16; ++i) {
        int c = ty * 16 + i;
        size_t nchw = ((size_t)(b * C_DIM + c)) * HW + hw;
        q[nchw] = w[(size_t)id * C_DIM + c];
        atomicAdd(&sums[(size_t)id * C_DIM + c], enc[nchw]);
    }
}

// ---------------------------------------------------------------------------
// Kernel 5: new_weight = 0.99*w + 0.01 * sums/(counts + 1e-12)
// ---------------------------------------------------------------------------
__global__ void vq_update_w(const float* __restrict__ w, const float* __restrict__ sums,
                            const float* __restrict__ counts, float* __restrict__ nw) {
    int tid = blockIdx.x * blockDim.x + threadIdx.x;
    if (tid >= K_CODES * C_DIM) return;
    int k = tid / C_DIM;
    float cnt = counts[k];
    nw[tid] = 0.99f * w[tid] + 0.01f * (sums[tid] / (cnt + 1e-12f));
}

// ---------------------------------------------------------------------------
extern "C" void kernel_launch(void* const* d_in, const int* in_sizes, int n_in,
                              void* d_out, int out_size, void* d_ws, size_t ws_size,
                              hipStream_t stream) {
    (void)in_sizes; (void)n_in; (void)out_size; (void)ws_size;
    const float* enc = (const float*)d_in[0];   // [16,128,32,32] f32
    const float* w   = (const float*)d_in[1];   // [8192,128]     f32

    // workspace layout
    char* ws = (char*)d_ws;
    _Float16* xh    = (_Float16*)ws;                                       // 4 MB
    _Float16* wh    = (_Float16*)(ws + (size_t)N_PTS * C_DIM * 2);         // 2 MB
    float*    wnorm = (float*)(ws + (size_t)N_PTS * C_DIM * 2
                                  + (size_t)K_CODES * C_DIM * 2);          // 32 KB
    float*    counts = wnorm + K_CODES;                                    // 32 KB
    float*    sums   = counts + K_CODES;                                   // 4 MB

    // output layout (flat, in return order): ids[N] (int bits), q[N*C], new_w[K*C]
    int*   ids = (int*)d_out;
    float* q   = (float*)d_out + N_PTS;
    float* nw  = q + (size_t)N_PTS * C_DIM;

    // zero EMA accumulators (counts + sums are contiguous)
    (void)hipMemsetAsync(counts, 0,
                         (size_t)(K_CODES + K_CODES * C_DIM) * sizeof(float), stream);

    vq_prep_x<<<(16 * 4 * 32), 256, 0, stream>>>(enc, xh);
    vq_prep_w<<<K_CODES, 128, 0, stream>>>(w, wh, wnorm);
    vq_argmin_wmma<<<N_PTS / 128, 256, 0, stream>>>(xh, wh, wnorm, ids);
    vq_scatter<<<N_PTS / 32, 256, 0, stream>>>(ids, w, enc, q, sums, counts);
    vq_update_w<<<(K_CODES * C_DIM + 255) / 256, 256, 0, stream>>>(w, sums, counts, nw);
}
